// CustomDense_6725918785732
// MI455X (gfx1250) — matrix operand order repaired and checked by reading, not compiled, not run on hardware
//
#include <hip/hip_runtime.h>
#include <stdint.h>

typedef __attribute__((ext_vector_type(16))) _Float16 v16h;
typedef __attribute__((ext_vector_type(8)))  _Float16 v8h;
typedef __attribute__((ext_vector_type(4)))  _Float16 v4h;
typedef __attribute__((ext_vector_type(8)))  float    v8f;
typedef __attribute__((ext_vector_type(4)))  float    v4f;

#define D_IN  2048
#define UNITS 8192
#define ROWS  8192

#define BM 256
#define BN 128
#define BK 32
#define SA 40   // LDS row stride (halfs): 80B = 20 dwords -> conflict-free b128 frag loads
#define SB 40
#define NK (D_IN / BK)   // 64 k-steps

static_assert(ROWS % BM == 0 && UNITS % BN == 0 && D_IN % BK == 0, "tiling");

__global__ __launch_bounds__(256)
void dense_i8_f16wmma_kernel(const float* __restrict__ Ag,
                             const int8_t* __restrict__ Wg,
                             const float* __restrict__ scaleP,
                             const float* __restrict__ zpP,
                             const float* __restrict__ biasG,
                             float* __restrict__ Out)
{
    __shared__ _Float16 lA[2][BM * SA];   // [m][k] f16, 20 KB per buffer
    __shared__ _Float16 lB[2][BN * SB];   // [n][k] f16 (transposed+dequantized), 10 KB per buffer

    const int tid  = threadIdx.x;
    const int lane = tid & 31;
    const int wave = tid >> 5;
    const int bm   = blockIdx.y * BM;
    const int bn   = blockIdx.x * BN;

    const float scale = scaleP[0];          // uniform -> s_load
    const float nzps  = -zpP[0] * scale;    // fused dequant: w*scale + nzps

    // ---- global->LDS fill mapping -------------------------------------
    // A: each thread owns one full row of the 256x32 tile (32 floats)
    const float* aptr = Ag + (size_t)(bm + tid) * D_IN;
    // W: each thread owns a 4k x 4n patch: 32 n-groups x 8 k-groups
    const int wn = (tid & 31) << 2;   // 0..124
    const int wk = (tid >> 5) << 2;   // 0..28
    const int8_t* wptr = Wg + (size_t)wk * UNITS + (bn + wn);

    v4f astg[8];     // 32 fp32 staged from A
    int wstg[4];     // 4x4 int8 staged from W

    auto gload = [&](int ks) {
        const float* ap = aptr + ks * BK;
#pragma unroll
        for (int q = 0; q < 8; ++q)
            astg[q] = *(const v4f*)(ap + 4 * q);
        const int8_t* wp = wptr + (size_t)ks * BK * UNITS;
#pragma unroll
        for (int i = 0; i < 4; ++i)
            wstg[i] = *(const int*)(wp + (size_t)i * UNITS);
    };

    auto lstore = [&](int buf) {
        // A: fp32 -> f16, four ds_store_b128
        _Float16* la = &lA[buf][tid * SA];
#pragma unroll
        for (int q = 0; q < 4; ++q) {
            v8h h;
#pragma unroll
            for (int j = 0; j < 4; ++j) {
                h[j]     = (_Float16)astg[2 * q][j];
                h[j + 4] = (_Float16)astg[2 * q + 1][j];
            }
            *(v8h*)(la + 8 * q) = h;
        }
        // W: dequant int8 -> f16 (fused fma), transpose 4x4 patch, four ds_store_b64
#pragma unroll
        for (int j = 0; j < 4; ++j) {          // n within patch
            v4h hv;
#pragma unroll
            for (int i = 0; i < 4; ++i) {      // k within patch
                int8_t sb = (int8_t)(wstg[i] >> (8 * j));
                hv[i] = (_Float16)__builtin_fmaf((float)sb, scale, nzps);
            }
            *(v4h*)(&lB[buf][(wn + j) * SB + wk]) = hv;
        }
    };

    // ---- per-wave compute mapping -------------------------------------
    const int wmO  = (wave & 3) * 64;   // wave tile M offset in block
    const int wnO  = (wave >> 2) * 64;  // wave tile N offset in block
    const int frow = lane & 15;
    const int fhi  = lane >> 4;         // 0: lanes 0-15, 1: lanes 16-31

    v8f acc[4][4] = {};                 // 4 m-tiles x 4 n-tiles of 16x16 f32

    auto compute = [&](int buf) {
        v16h afrag[4], bfrag[4];
        // A frag (16x32 f16): lane-low V0-3=K0-7,V4-7=K16-23; lane-high K8-15,K24-31
        const int akh = fhi << 3;       // 0 or 8 halfs
#pragma unroll
        for (int tm = 0; tm < 4; ++tm) {
            const _Float16* p = &lA[buf][(wmO + tm * 16 + frow) * SA];
            v8h lo = *(const v8h*)(p + akh);
            v8h hi = *(const v8h*)(p + 16 + akh);
            afrag[tm] = __builtin_shufflevector(lo, hi, 0,1,2,3,4,5,6,7,8,9,10,11,12,13,14,15);
        }
        // B frag (32x16 f16): lane-low V0-3=K0-7,V4-7=K16-23; lane-high K8-15,K24-31
        const int bkh = fhi << 3;       // 0 or 8 halfs
#pragma unroll
        for (int tn = 0; tn < 4; ++tn) {
            const _Float16* p = &lB[buf][(wnO + tn * 16 + frow) * SB + bkh];
            v8h lo = *(const v8h*)(p);
            v8h hi = *(const v8h*)(p + 16);
            bfrag[tn] = __builtin_shufflevector(lo, hi, 0,1,2,3,4,5,6,7,8,9,10,11,12,13,14,15);
        }
#pragma unroll
        for (int tm = 0; tm < 4; ++tm)
#pragma unroll
            for (int tn = 0; tn < 4; ++tn)
                acc[tm][tn] = __builtin_amdgcn_wmma_f32_16x16x32_f16(
                    false, afrag[tm], false, bfrag[tn],
                    (short)0, acc[tm][tn], false, false);
    };

    // ---- pipelined main loop ------------------------------------------
    gload(0);
    lstore(0);
    __syncthreads();
#pragma unroll 1
    for (int ks = 0; ks < NK; ++ks) {
        if (ks + 1 < NK) gload(ks + 1);     // issue next-stage global loads
        compute(ks & 1);                    // 16 WMMAs hide the load latency
        if (ks + 1 < NK) lstore((ks + 1) & 1);
        __syncthreads();
    }

    // ---- epilogue: bias + relu + store --------------------------------
#pragma unroll
    for (int tn = 0; tn < 4; ++tn) {
        const int col = bn + wnO + tn * 16 + frow;
        const float bv = biasG[col];
#pragma unroll
        for (int tm = 0; tm < 4; ++tm) {
            const int r0 = bm + wmO + tm * 16 + (fhi << 3);
#pragma unroll
            for (int r = 0; r < 8; ++r) {
                float v = acc[tm][tn][r] + bv;
                Out[(size_t)(r0 + r) * UNITS + col] = v > 0.f ? v : 0.f;
            }
        }
    }
}

extern "C" void kernel_launch(void* const* d_in, const int* in_sizes, int n_in,
                              void* d_out, int out_size, void* d_ws, size_t ws_size,
                              hipStream_t stream) {
    const float*  A     = (const float*)d_in[0];
    const int8_t* W     = (const int8_t*)d_in[1];
    const float*  scale = (const float*)d_in[2];
    const float*  zp    = (const float*)d_in[3];
    const float*  bias  = (const float*)d_in[4];
    float* out = (float*)d_out;

    dim3 grid(UNITS / BN, ROWS / BM);   // (64, 32)
    dim3 block(256);
    dense_i8_f16wmma_kernel<<<grid, block, 0, stream>>>(A, W, scale, zp, bias, out);
}
